// HypAgg_57286273794079
// MI455X (gfx1250) — matrix operand order repaired and checked by reading, not compiled, not run on hardware
//
#include <hip/hip_runtime.h>

typedef unsigned int u32;
typedef unsigned short u16;
typedef unsigned long long u64;
typedef __attribute__((ext_vector_type(4)))  u32    u32x4;
typedef __attribute__((ext_vector_type(2)))  float  f32x2;
typedef __attribute__((ext_vector_type(16))) __bf16 v16bf;
typedef __attribute__((ext_vector_type(8)))  float  v8f;

#define N_ROWS 8192
#define NF     256

// float -> bf16 round-to-nearest-even (values are finite; skip NaN special-case)
static __device__ __forceinline__ u32 f2bf(float f) {
    u32 x = __float_as_uint(f);
    return (x + 0x7fffu + ((x >> 16) & 1u)) >> 16;
}

// fast sigmoid: single v_exp_f32 + single v_rcp_f32 (result is rounded to bf16
// anyway, so ~1ulp rcp is plenty)
static __device__ __forceinline__ float fast_sigmoid(float s) {
    return __builtin_amdgcn_rcpf(1.0f + __expf(-s));
}

union Frag16 { u32x4 q[2]; v16bf v; };

// ---------------------------------------------------------------------------
// Kernel 1: logmap0(x) -> xt ; sl = xt.w[:F] ; sr = xt.w[F:] ; store xtT bf16
// One wave per row, 8 rows per 256-thread block.
// ---------------------------------------------------------------------------
__global__ void __launch_bounds__(256)
k_logmap_scores(const float* __restrict__ x,
                const float* __restrict__ w_att,
                u16*   __restrict__ xtT,     // [NF][N_ROWS] bf16 (transposed)
                float* __restrict__ sl,
                float* __restrict__ sr)
{
    __shared__ u16 ldsT[NF * 8];             // [f][row_in_block]

    const int tid  = threadIdx.x;
    const int w    = tid >> 5;               // wave id = row in block (0..7)
    const int lane = tid & 31;
    const int i    = blockIdx.x * 8 + w;

    // load 8 consecutive floats per lane: f = lane*8 + k
    const float4* xrow = reinterpret_cast<const float4*>(x + (size_t)i * NF);
    float4 p0 = xrow[lane * 2];
    float4 p1 = xrow[lane * 2 + 1];
    float v[8] = {p0.x, p0.y, p0.z, p0.w, p1.x, p1.y, p1.z, p1.w};

    // sum of squares over f >= 1
    float ss = 0.f;
    #pragma unroll
    for (int k = 0; k < 8; ++k) {
        float t = v[k];
        if (!(lane == 0 && k == 0)) ss += t * t;
    }
    #pragma unroll
    for (int m = 16; m > 0; m >>= 1) ss += __shfl_xor(ss, m, 32);

    float x0    = __shfl(v[0], 0, 32);
    float ynorm = fmaxf(sqrtf(ss), 1e-15f);
    float theta = fmaxf(x0, 1.0f + 1e-7f);
    // arccosh(theta) = log(theta + sqrt(theta^2 - 1))
    float scale = __logf(theta + sqrtf(theta * theta - 1.0f)) *
                  __builtin_amdgcn_rcpf(ynorm);

    float xt[8];
    float dl = 0.f, dr = 0.f;
    #pragma unroll
    for (int k = 0; k < 8; ++k) {
        int f = lane * 8 + k;
        float t = scale * v[k];
        if (lane == 0 && k == 0) t = 0.f;    // xt[:,0] = 0
        xt[k] = t;
        dl += t * w_att[f];
        dr += t * w_att[NF + f];
    }
    #pragma unroll
    for (int m = 16; m > 0; m >>= 1) {
        dl += __shfl_xor(dl, m, 32);
        dr += __shfl_xor(dr, m, 32);
    }
    if (lane == 0) { sl[i] = dl; sr[i] = dr; }

    // transpose through LDS: ldsT[f][w]
    #pragma unroll
    for (int k = 0; k < 8; ++k)
        ldsT[(lane * 8 + k) * 8 + w] = (u16)f2bf(xt[k]);
    __syncthreads();

    // thread t writes row f=t of xtT: 8 bf16 = 16 bytes, contiguous
    int f = tid;
    u32x4 rowq = *reinterpret_cast<const u32x4*>(&ldsT[f * 8]);
    *reinterpret_cast<u32x4*>(&xtT[(size_t)f * N_ROWS + blockIdx.x * 8]) = rowq;
}

// ---------------------------------------------------------------------------
// Kernel 2: fused  att = sigmoid(sl_i + sr_j + b) * adj  ;  support = att @ xt
//           ;  out = proj(expmap0(support))
// Block: 512 threads = 16 waves. Block owns 16 i-rows x all 256 f-cols.
// Wave w produces f-tile [w*16, w*16+16). K-loop over j in steps of 64:
// each thread makes 2 att elements (consecutive K pair, packed b32 store),
// each wave issues 2 WMMAs per barrier.
// ---------------------------------------------------------------------------
__global__ void __launch_bounds__(512)
k_agg(const float* __restrict__ adj,
      const u16*   __restrict__ xtT,     // [NF][N_ROWS] bf16
      const float* __restrict__ sl,
      const float* __restrict__ sr,
      const float* __restrict__ b_att,
      float* __restrict__ out)
{
    // LDS: during K-loop -> double-buffered bf16 A tiles [2][2][16][40] (5 KB)
    //      after loop    -> f32 supp tile [16][260] (16.6 KB)
    __shared__ __align__(16) unsigned char smem[16 * 260 * 4];
    u16*   attb = reinterpret_cast<u16*>(smem);
    float* supp = reinterpret_cast<float*>(smem);

    const int tid  = threadIdx.x;
    const int w    = tid >> 5;               // wave id 0..15
    const int lane = tid & 31;
    const int i0   = blockIdx.x * 16;

    const float bb   = b_att[0];
    const float sl_w = sl[i0 + w];           // wave w computes att row M=w

    // att production: thread owns K pair {2*lane, 2*lane+1} of the 64-strip;
    // lanes 0-15 -> tile 0 (K 0..31), lanes 16-31 -> tile 1 (K 32..63)
    const int atile = (lane >> 4);                  // 0 or 1
    const int acol  = (lane * 2) & 31;              // even column
    u16* astore = attb; // + buf*1280 later
    const int astoff = atile * 640 + w * 40 + acol; // halfs, 4B aligned

    // A-fragment addressing (16-bit A 16x32 layout):
    const int arow = lane & 15;
    const int akb  = (lane < 16) ? 0 : 8;
    // B-fragment addressing (bf16 B 32x16 layout):
    const int fcol = w * 16 + (lane & 15);
    const int koff = (lane < 16) ? 0 : 8;
    const u16* xbase = xtT + (size_t)fcol * N_ROWS + koff;

    const float* adjrow = adj + (size_t)(i0 + w) * N_ROWS;

    v8f acc = {};

    for (int k0 = 0; k0 < N_ROWS; k0 += 64) {
        const int buf = (k0 >> 6) & 1;
        u16* abuf = astore + buf * 1280;     // 2 tiles of 16x40 halfs

        // --- produce att K-pair (M=w, K = k0 + 2*lane, +1) ---
        f32x2 a2 = *reinterpret_cast<const f32x2*>(adjrow + k0 + 2 * lane);
        f32x2 r2 = *reinterpret_cast<const f32x2*>(sr + k0 + 2 * lane);
        float s0 = sl_w + r2.x + bb;
        float s1 = sl_w + r2.y + bb;
        u32 packed = f2bf(a2.x * fast_sigmoid(s0)) |
                     (f2bf(a2.y * fast_sigmoid(s1)) << 16);
        *reinterpret_cast<u32*>(abuf + astoff) = packed;
        __syncthreads();

        // --- tile 0: K = k0 .. k0+31 ---
        Frag16 af0, bf0;
        const u16* ap0 = abuf + arow * 40 + akb;
        af0.q[0] = *reinterpret_cast<const u32x4*>(ap0);
        af0.q[1] = *reinterpret_cast<const u32x4*>(ap0 + 16);
        const u16* bp0 = xbase + k0;
        bf0.q[0] = *reinterpret_cast<const u32x4*>(bp0);
        bf0.q[1] = *reinterpret_cast<const u32x4*>(bp0 + 16);
        acc = __builtin_amdgcn_wmma_f32_16x16x32_bf16(
                  false, af0.v, false, bf0.v, (short)0, acc, false, false);

        // --- tile 1: K = k0+32 .. k0+63 ---
        Frag16 af1, bf1;
        const u16* ap1 = abuf + 640 + arow * 40 + akb;
        af1.q[0] = *reinterpret_cast<const u32x4*>(ap1);
        af1.q[1] = *reinterpret_cast<const u32x4*>(ap1 + 16);
        const u16* bp1 = xbase + k0 + 32;
        bf1.q[0] = *reinterpret_cast<const u32x4*>(bp1);
        bf1.q[1] = *reinterpret_cast<const u32x4*>(bp1 + 16);
        acc = __builtin_amdgcn_wmma_f32_16x16x32_bf16(
                  false, af1.v, false, bf1.v, (short)0, acc, false, false);
    }
    __syncthreads();

    // --- scatter accumulators into supp[16][260] (C layout: VGPR r, lane l:
    //     row = (l<16 ? r : r+8), col = w*16 + l%16) ---
    {
        const int col   = w * 16 + (lane & 15);
        const int rbase = (lane < 16) ? 0 : 8;
        #pragma unroll
        for (int r = 0; r < 8; ++r)
            supp[(rbase + r) * 260 + col] = acc[r];
    }
    __syncthreads();

    // --- epilogue: wave w handles output row i0+w ---
    const float* srow = supp + w * 260 + lane * 8;
    float vals[8];
    float ss = 0.f;
    #pragma unroll
    for (int k = 0; k < 8; ++k) {
        float t = srow[k];
        vals[k] = t;
        if (!(lane == 0 && k == 0)) ss += t * t;   // exclude u[...,0]
    }
    #pragma unroll
    for (int m = 16; m > 0; m >>= 1) ss += __shfl_xor(ss, m, 32);

    float norm  = fmaxf(sqrtf(ss), 1e-15f);
    float theta = norm;                       // sqrtK = 1
    float e     = __expf(theta);
    float einv  = __builtin_amdgcn_rcpf(e);
    float sh    = 0.5f * (e - einv);
    float scale = sh * __builtin_amdgcn_rcpf(norm);     // y = scale * u
    float x0    = sqrtf(1.0f + scale * scale * ss);     // proj: sqrt(K + sum y^2)

    float* orow = out + (size_t)(i0 + w) * NF + lane * 8;
    #pragma unroll
    for (int k = 0; k < 8; ++k) orow[k] = scale * vals[k];
    if (lane == 0) orow[0] = x0;
}

// ---------------------------------------------------------------------------
extern "C" void kernel_launch(void* const* d_in, const int* in_sizes, int n_in,
                              void* d_out, int out_size, void* d_ws, size_t ws_size,
                              hipStream_t stream) {
    const float* x     = (const float*)d_in[0];   // (8192, 256)
    const float* adj   = (const float*)d_in[1];   // (8192, 8192)
    const float* w_att = (const float*)d_in[2];   // (512,)
    const float* b_att = (const float*)d_in[3];   // (1,)
    float* out = (float*)d_out;                   // (8192, 256)

    // workspace layout
    u16*   xtT = (u16*)d_ws;                                    // 4 MB
    float* sl  = (float*)((char*)d_ws + (size_t)NF * N_ROWS * 2);
    float* sr  = sl + N_ROWS;

    k_logmap_scores<<<N_ROWS / 8, 256, 0, stream>>>(x, w_att, xtT, sl, sr);
    k_agg<<<N_ROWS / 16, 512, 0, stream>>>(adj, xtT, sl, sr, b_att, out);
}
